// RNNG_45904610459911
// MI455X (gfx1250) — compile-verified, matrix operand order checked by reading, not run on hardware
//
#include <hip/hip_runtime.h>
#include <hip/hip_bf16.h>
#include <cstdint>

// RNNG forward on MI455X (gfx1250): single persistent workgroup (8 wave32 waves).
// All GEMMs via v_wmma_f32_16x16x32_f16 (f32 accumulate). Weights pre-packed into
// WMMA B-fragment tiles (one 32B global_load per lane), batch-loaded into distinct
// register buffers ahead of each WMMA chain so L2 latency overlaps matrix issue.
// Matmul A-operands staged through LDS (ds_load_b128). Stacks stored f16.
// All global workspace traffic is strictly column-partitioned per thread (no
// cross-thread global communication -> no volatile, no fences).

#define B_   32
#define S_   32
#define H_   256
#define V_   12000
#define T_   31
#define H3_  768

typedef _Float16 v16h __attribute__((ext_vector_type(16)));
typedef _Float16 v8h  __attribute__((ext_vector_type(8)));
typedef float    v8f  __attribute__((ext_vector_type(8)));
typedef float    v4f  __attribute__((ext_vector_type(4)));

__device__ __forceinline__ v8f splat8(float v){
  v8f r;
#pragma unroll
  for (int i = 0; i < 8; ++i) r[i] = v;
  return r;
}

__device__ __forceinline__ v8f wmma_f16(v16h a, v16h b, v8f c){
  return __builtin_amdgcn_wmma_f32_16x16x32_f16(false, a, false, b, (short)0, c, false, false);
}

// 16-bit A 16x32 layout: lane L holds row m0+(L&15); K chunks [h2*8 .. +7] and
// [16+h2*8 .. +7] where h2 = L>>4  ->  two contiguous 16B LDS loads.
__device__ __forceinline__ v16h afrag_lds16(const _Float16* x, int stride,
                                            int m0, int k0, int lane){
  const _Float16* row = x + (m0 + (lane & 15)) * stride + k0 + ((lane >> 4) << 3);
  v8h lo = *(const v8h*)row;
  v8h hi = *(const v8h*)(row + 16);
  v16h a;
#pragma unroll
  for (int i = 0; i < 8; ++i){ a[i] = lo[i]; a[8 + i] = hi[i]; }
  return a;
}

__device__ __forceinline__ v16h afrag_lds32(const float* x, int stride,
                                            int m0, int k0, int lane){
  const float* row = x + (m0 + (lane & 15)) * stride + k0 + ((lane >> 4) << 3);
  v4f a0 = *(const v4f*)row,        a1 = *(const v4f*)(row + 4);
  v4f b0 = *(const v4f*)(row + 16), b1 = *(const v4f*)(row + 20);
  v16h a;
#pragma unroll
  for (int i = 0; i < 4; ++i){
    a[i]      = (_Float16)a0[i];
    a[4 + i]  = (_Float16)a1[i];
    a[8 + i]  = (_Float16)b0[i];
    a[12 + i] = (_Float16)b1[i];
  }
  return a;
}

// Packed B tiles: tile (ntile, ktile) = 512 halfs; lane reads one 32B v16h.
__device__ __forceinline__ v16h bfrag(const _Float16* __restrict__ pk,
                                      int ktile, int ntile, int KT, int lane){
  const v16h* p = (const v16h*)(pk + (((size_t)(ntile * KT + ktile)) << 9));
  return p[lane];
}

__device__ __forceinline__ float sigf(float x){ return 1.f / (1.f + __expf(-x)); }

__device__ __forceinline__ int kmap(int j, int h2){
  return ((j < 4) ? (j << 1) : (16 + ((j - 4) << 1))) + (h2 << 3);
}

// h = mask ? gru(x, h) : h.  x: LDS f16 [32][256], h: LDS f32 [32][256].
// Each wave owns 2 m-tiles x 2 col-tiles; gates live in registers only.
// Per k-step: batch all 6 B-fragment loads (one clause), then 6 WMMAs.
__device__ __forceinline__ void gru_step(const _Float16* x, float* h,
                                         const _Float16* __restrict__ Wpk,
                                         const _Float16* __restrict__ Upk,
                                         const float* __restrict__ bias,
                                         const int* mask)
{
  const int tid = threadIdx.x, wave = tid >> 5, lane = tid & 31;
  v8f keep[4]; int rowb[4], colv[4];
#pragma unroll
  for (int q = 0; q < 4; ++q){
    const int m0  = (q >> 1) << 4;
    const int c   = (wave << 1) + (q & 1);
    const int col = (c << 4) + (lane & 15);
    v8f az  = splat8(bias[col]);
    v8f ar  = splat8(bias[H_ + col]);
    v8f axn = splat8(bias[2 * H_ + col]);
    v8f ahn = splat8(0.f);
#pragma unroll
    for (int kt = 0; kt < 8; ++kt){
      // batch loads into distinct buffers -> deep load pipeline
      v16h bwz = bfrag(Wpk, kt, c,      8, lane);
      v16h buz = bfrag(Upk, kt, c,      8, lane);
      v16h bwr = bfrag(Wpk, kt, 16 + c, 8, lane);
      v16h bur = bfrag(Upk, kt, 16 + c, 8, lane);
      v16h bwn = bfrag(Wpk, kt, 32 + c, 8, lane);
      v16h bun = bfrag(Upk, kt, 32 + c, 8, lane);
      v16h ax  = afrag_lds16(x, H_, m0, kt << 5, lane);
      v16h ah  = afrag_lds32(h, H_, m0, kt << 5, lane);
      az  = wmma_f16(ax, bwz, az);
      az  = wmma_f16(ah, buz, az);
      ar  = wmma_f16(ax, bwr, ar);
      ar  = wmma_f16(ah, bur, ar);
      axn = wmma_f16(ax, bwn, axn);
      ahn = wmma_f16(ah, bun, ahn);
    }
    const int mr = m0 + ((lane >> 4) << 3);
    v8f o;
#pragma unroll
    for (int i = 0; i < 8; ++i){
      float z = sigf(az[i]);
      float r = sigf(ar[i]);
      float n = tanhf(axn[i] + r * ahn[i]);
      float ho = h[(mr + i) * H_ + col];
      o[i] = (1.f - z) * ho + z * n;
    }
    keep[q] = o; rowb[q] = mr; colv[q] = col;
  }
  __syncthreads();                       // all h reads done before writes
#pragma unroll
  for (int q = 0; q < 4; ++q)
#pragma unroll
    for (int i = 0; i < 8; ++i){
      int m = rowb[q] + i;
      if (!mask || mask[m] > 0) h[m * H_ + colv[q]] = keep[q][i];
    }
  __syncthreads();
}

// attacc[q] += x(LDS f16 [32][256]) @ Wc[ktBase*32 : +256, :], per-wave tiles.
// All 8 B-fragments preloaded per tile before the WMMA chain.
__device__ __forceinline__ void att_accum(v8f* acc, const _Float16* x,
                                          const _Float16* __restrict__ WcP,
                                          int ktBase)
{
  const int tid = threadIdx.x, wave = tid >> 5, lane = tid & 31;
#pragma unroll
  for (int q = 0; q < 4; ++q){
    const int m0 = (q >> 1) << 4;
    const int c  = (wave << 1) + (q & 1);
    v16h bfr[8];
#pragma unroll
    for (int kt = 0; kt < 8; ++kt)
      bfr[kt] = bfrag(WcP, ktBase + kt, c, 24, lane);
#pragma unroll
    for (int kt = 0; kt < 8; ++kt){
      v16h a = afrag_lds16(x, H_, m0, kt << 5, lane);
      acc[q] = wmma_f16(a, bfr[kt], acc[q]);
    }
  }
}

// f32 weight [K,N] row-major -> f16 WMMA B-fragment tiles.
__global__ void pack_weights(const float* __restrict__ W, _Float16* __restrict__ out,
                             int K, int N)
{
  int idx = blockIdx.x * blockDim.x + threadIdx.x;
  if (idx >= K * N) return;
  int e    =  idx        & 1;
  int j    = (idx >> 1)  & 7;
  int lane = (idx >> 4)  & 31;
  int tile =  idx >> 9;
  int KT = K >> 5;
  int ktile = tile % KT, ntile = tile / KT;
  int n = (ntile << 4) + (lane & 15);
  int k = (ktile << 5) + kmap(j, lane >> 4) + e;
  out[idx] = (_Float16)W[k * N + n];
}

__global__ void __launch_bounds__(256)
rnng_main(const int* __restrict__ actions, const float* __restrict__ embed,
          const float* __restrict__ ab, const float* __restrict__ bb,
          const float* __restrict__ sb, const float* __restrict__ rb,
          const float* __restrict__ bc, const float* __restrict__ bp,
          const _Float16* __restrict__ aWp, const _Float16* __restrict__ aUp,
          const _Float16* __restrict__ bWp, const _Float16* __restrict__ bUp,
          const _Float16* __restrict__ sWp, const _Float16* __restrict__ sUp,
          const _Float16* __restrict__ rWp, const _Float16* __restrict__ rUp,
          const _Float16* __restrict__ WcP, const _Float16* __restrict__ WpP,
          _Float16* __restrict__ sm, _Float16* __restrict__ bm, int* __restrict__ tm,
          float* __restrict__ ahw, float* __restrict__ out)
{
  __shared__ __align__(16) float    hbuf[B_ * H_];   // scan hidden state (f32)
  __shared__ __align__(16) _Float16 xcur[B_ * H_];   // staged x / att (f16)
  __shared__ int st[B_], tt[B_], bt[B_], accv[B_], laA[B_], isred[B_], irv[B_], av[B_];
  __shared__ int maskArr[B_], pdo[B_], pidx[B_];
  __shared__ int ntdo[B_], ntidx[B_], gdo[B_], gidx[B_], bdo[B_], bidx[B_];
  __shared__ float wbest[8][B_];
  __shared__ int   widx[8][B_];
  __shared__ int anyRed, maskSum, redActive, redGo, encTop;

  const int tid = threadIdx.x, wave = tid >> 5, lane = tid & 31;

  if (tid < B_){ st[tid] = -1; tt[tid] = -1; bt[tid] = 0; accv[tid] = 1; }
  for (int e = tid; e < B_ * H_; e += 256) ahw[e] = 0.f;          // ah = h0
  for (int e = tid; e < B_ * H_; e += 256){                       // bm[:,0]=embed[START]
    int b = e >> 8, c = e & 255;
    bm[(size_t)(b * S_) * H_ + c] = (_Float16)embed[1 * H_ + c];
  }
  __syncthreads();

  for (int t = 0; t < T_; ++t){
    // ---- action decode ----
    if (tid < B_){
      int a = actions[tid * S_ + t];
      int l = a * accv[tid];
      laA[tid]   = l;
      isred[tid] = (l == 2) ? 1 : 0;
      irv[tid]   = isred[tid];
      av[tid]    = 1;
    }
    if (tid == 0) maskSum = 0;
    __syncthreads();

    // ---- REDUCE: masked pop+GRU, early-exit when all ir counters die ----
    for (int e = tid; e < B_ * H_; e += 256) hbuf[e] = 0.f;       // h0
    __syncthreads();
    for (int j = 0; j < S_; ++j){
      if (tid < B_){
        int b = tid;
        int ira  = irv[b] * av[b];
        int can1 = (st[b] >= 0) ? 1 : 0;
        int do1  = (ira > 0) && can1;
        pdo[b]  = do1;
        pidx[b] = do1 ? st[b] : 0;
        if (do1) st[b]--;
        av[b] *= (ira > 0) ? can1 : 1;
        int ira2 = irv[b] * av[b];
        int can2 = (tt[b] >= 0) ? 1 : 0;
        int do2  = (ira2 > 0) && can2;
        int tag  = do2 ? tm[b * S_ + tt[b]] : 0;
        if (do2) tt[b]--;
        av[b] *= (ira2 > 0) ? can2 : 1;
        int outm = irv[b] * av[b];
        maskArr[b] = outm;
        irv[b] = (irv[b] - tag) * av[b];
        if (outm > 0) atomicAdd(&maskSum, 1);
      }
      __syncthreads();
      if (tid == 0){
        int go = 0, act = 0;
        for (int b = 0; b < B_; ++b){
          go  |= (maskArr[b] > 0);
          act |= (irv[b] * av[b] > 0);
        }
        redGo = go; redActive = act;
      }
      __syncthreads();
      if (redGo){
        for (int e = tid; e < B_ * H_; e += 256){
          int b = e >> 8, c = e & 255;
          xcur[e] = pdo[b] ? sm[(size_t)(b * S_ + pidx[b]) * H_ + c]
                           : (_Float16)0.f;
        }
        __syncthreads();
        gru_step(xcur, hbuf, rWp, rUp, rb, maskArr);
      }
      if (!redActive) break;
    }
    // push rfinal (hbuf), then tag 0
    if (tid < B_){
      int b = tid;
      int m1  = isred[b] * av[b];
      int can = (st[b] + 1 < S_) ? 1 : 0;
      int d   = (m1 > 0) && can;
      pdo[b]  = d;
      pidx[b] = d ? st[b] + 1 : 0;
      if (d) st[b]++;
      av[b] *= (m1 > 0) ? can : 1;
      int m2   = isred[b] * av[b];
      int can2 = (tt[b] + 1 < S_) ? 1 : 0;
      int d2   = (m2 > 0) && can2;
      if (d2){ tt[b]++; tm[b * S_ + tt[b]] = 0; }
      av[b] *= (m2 > 0) ? can2 : 1;
    }
    __syncthreads();
    for (int e = tid; e < B_ * H_; e += 256){
      int b = e >> 8, c = e & 255;
      if (pdo[b]) sm[(size_t)(b * S_ + pidx[b]) * H_ + c] = (_Float16)hbuf[b * H_ + c];
    }
    if (tid == 0){
      int any = 0;
      for (int b = 0; b < B_; ++b) any |= (isred[b] > 0);
      anyRed = any;
    }
    __syncthreads();
    if (tid < B_){
      int rm = anyRed ? ((maskSum == 0) ? 0 : av[tid]) : 1;
      accv[tid] *= rm;
    }
    __syncthreads();

    // ---- NT / GEN pushes ----
    if (tid < B_){
      int b = tid, l = laA[b];
      int isnt = (l > 2 && l < 16) ? 1 : 0;
      int isg  = (l >= 16) ? 1 : 0;
      int can5 = (st[b] + 1 < S_) ? 1 : 0;
      int d5   = isnt && can5;
      int s5   = isnt ? can5 : 1;
      ntdo[b] = d5; ntidx[b] = d5 ? st[b] + 1 : 0; if (d5) st[b]++;
      int m6 = isnt * s5;
      int can6 = (tt[b] + 1 < S_) ? 1 : 0;
      int d6 = (m6 > 0) && can6;
      if (d6){ tt[b]++; tm[b * S_ + tt[b]] = 1; }
      int s6 = (m6 > 0) ? can6 : 1;
      accv[b] *= s5 * s6;
      int can7 = (st[b] + 1 < S_) ? 1 : 0;
      int d7 = isg && can7;
      int s7 = isg ? can7 : 1;
      gdo[b] = d7; gidx[b] = d7 ? st[b] + 1 : 0; if (d7) st[b]++;
      int m8 = isg * s7;
      int can8 = (tt[b] + 1 < S_) ? 1 : 0;
      int d8 = (m8 > 0) && can8;
      if (d8){ tt[b]++; tm[b * S_ + tt[b]] = 0; }
      int s8 = (m8 > 0) ? can8 : 1;
      int m9 = isg * s7 * s8;
      int can9 = (bt[b] + 1 < S_) ? 1 : 0;
      int d9 = (m9 > 0) && can9;
      bdo[b] = d9; bidx[b] = d9 ? bt[b] + 1 : 0; if (d9) bt[b]++;
      int s9 = (m9 > 0) ? can9 : 1;
      accv[b] *= s7 * s8 * s9;
    }
    __syncthreads();
    for (int e = tid; e < B_ * H_; e += 256){
      int b = e >> 8, c = e & 255;
      _Float16 ev = (_Float16)embed[laA[b] * H_ + c];
      if (ntdo[b]) sm[(size_t)(b * S_ + ntidx[b]) * H_ + c] = ev;
      if (gdo[b])  sm[(size_t)(b * S_ + gidx[b])  * H_ + c] = ev;
      if (bdo[b])  bm[(size_t)(b * S_ + bidx[b])  * H_ + c] = ev;
    }
    __syncthreads();

    // per-wave attention accumulators (live only through encodes + att)
    v8f attacc[4];
#pragma unroll
    for (int q = 0; q < 4; ++q)
      attacc[q] = splat8(bc[(((wave << 1) + (q & 1)) << 4) + (lane & 15)]);

    // ---- buffer encode -> attacc (Wc rows 512:768, kt base 16) ----
    if (tid == 0){
      int mt = -1;
      for (int b = 0; b < B_; ++b) mt = (bt[b] > mt) ? bt[b] : mt;
      encTop = mt;
    }
    for (int e = tid; e < B_ * H_; e += 256) hbuf[e] = 0.f;
    __syncthreads();
    for (int s = 0; s <= encTop; ++s){
      if (tid < B_) maskArr[tid] = (s <= bt[tid]) ? 1 : 0;
      for (int e = tid; e < B_ * H_; e += 256){
        int b = e >> 8, c = e & 255;
        xcur[e] = bm[(size_t)(b * S_ + s) * H_ + c];
      }
      __syncthreads();
      gru_step(xcur, hbuf, bWp, bUp, bb, maskArr);
    }
    for (int e = tid; e < B_ * H_; e += 256) xcur[e] = (_Float16)hbuf[e];
    __syncthreads();
    att_accum(attacc, xcur, WcP, 16);
    __syncthreads();

    // ---- stack encode -> attacc (Wc rows 0:256, kt base 0) ----
    if (tid == 0){
      int mt = -1;
      for (int b = 0; b < B_; ++b) mt = (st[b] > mt) ? st[b] : mt;
      encTop = mt;
    }
    for (int e = tid; e < B_ * H_; e += 256) hbuf[e] = 0.f;
    __syncthreads();
    for (int s = 0; s <= encTop; ++s){
      if (tid < B_) maskArr[tid] = (s <= st[tid]) ? 1 : 0;
      for (int e = tid; e < B_ * H_; e += 256){
        int b = e >> 8, c = e & 255;
        xcur[e] = sm[(size_t)(b * S_ + s) * H_ + c];
      }
      __syncthreads();
      gru_step(xcur, hbuf, sWp, sUp, sb, maskArr);
    }
    for (int e = tid; e < B_ * H_; e += 256) xcur[e] = (_Float16)hbuf[e];
    __syncthreads();
    att_accum(attacc, xcur, WcP, 0);
    __syncthreads();

    // ---- action GRU -> attacc (Wc rows 256:512, kt base 8) ----
    for (int e = tid; e < B_ * H_; e += 256){
      int b = e >> 8, c = e & 255;
      xcur[e] = (_Float16)embed[laA[b] * H_ + c];   // emb_a
      hbuf[e] = ahw[e];
    }
    __syncthreads();
    gru_step(xcur, hbuf, aWp, aUp, ab, nullptr);
    for (int e = tid; e < B_ * H_; e += 256){
      ahw[e] = hbuf[e];
      xcur[e] = (_Float16)hbuf[e];
    }
    __syncthreads();
    att_accum(attacc, xcur, WcP, 8);
    __syncthreads();                                 // xcur reads done

    // ---- att = tanh(attacc) -> xcur (f16) ----
#pragma unroll
    for (int q = 0; q < 4; ++q){
      const int m0  = (q >> 1) << 4;
      const int col = (((wave << 1) + (q & 1)) << 4) + (lane & 15);
      const int mr  = m0 + ((lane >> 4) << 3);
#pragma unroll
      for (int i = 0; i < 8; ++i)
        xcur[(mr + i) * H_ + col] = (_Float16)tanhf(attacc[q][i]);
    }
    __syncthreads();

    // ---- logits + register argmax (per m-tile pass: lower VGPR pressure) ----
#pragma unroll 1
    for (int mi = 0; mi < 2; ++mi){
      float abest[8]; int aidx[8];
#pragma unroll
      for (int i = 0; i < 8; ++i){ abest[i] = -3.402823466e38f; aidx[i] = 0; }
      v16h afr[8];
#pragma unroll
      for (int kt = 0; kt < 8; ++kt)
        afr[kt] = afrag_lds16(xcur, H_, mi << 4, kt << 5, lane);
      for (int nt = wave; nt < V_ / 16; nt += 8){
        const int col = (nt << 4) + (lane & 15);
        v16h bfr[8];
#pragma unroll
        for (int kt = 0; kt < 8; ++kt)
          bfr[kt] = bfrag(WpP, kt, nt, 8, lane);
        v8f acc = splat8(bp[col]);
#pragma unroll
        for (int kt = 0; kt < 8; ++kt)
          acc = wmma_f16(afr[kt], bfr[kt], acc);
        const int mr = (mi << 4) + ((lane >> 4) << 3);
#pragma unroll
        for (int i = 0; i < 8; ++i){
          int b = mr + i;
          float v = acc[i];
          out[(size_t)(B_ * T_) + (size_t)(b * T_ + t) * V_ + col] = v;
          if (v > abest[i]){ abest[i] = v; aidx[i] = col; }
        }
      }
      // cross-lane argmax (lanes 0-15 / 16-31 hold disjoint row groups)
#pragma unroll
      for (int i = 0; i < 8; ++i){
        float bv = abest[i]; int bix = aidx[i];
#pragma unroll
        for (int m = 1; m < 16; m <<= 1){
          float ov = __shfl_xor(bv, m, 32);
          int   oi = __shfl_xor(bix, m, 32);
          if (ov > bv || (ov == bv && oi < bix)){ bv = ov; bix = oi; }
        }
        if ((lane & 15) == 0){
          int row = (mi << 4) + ((lane >> 4) << 3) + i;
          wbest[wave][row] = bv; widx[wave][row] = bix;
        }
      }
    }
    __syncthreads();
    if (tid < B_){
      float bv = wbest[0][tid]; int bix = widx[0][tid];
#pragma unroll
      for (int w = 1; w < 8; ++w){
        float ov = wbest[w][tid]; int oi = widx[w][tid];
        if (ov > bv || (ov == bv && oi < bix)){ bv = ov; bix = oi; }
      }
      out[tid * T_ + t] = (float)bix;
    }
    __syncthreads();
  }
}

extern "C" void kernel_launch(void* const* d_in, const int* in_sizes, int n_in,
                              void* d_out, int out_size, void* d_ws, size_t ws_size,
                              hipStream_t stream)
{
  const int*   actions = (const int*)  d_in[0];
  const float* embed   = (const float*)d_in[1];
  const float* aW = (const float*)d_in[2],  *aU = (const float*)d_in[3],  *ab = (const float*)d_in[4];
  const float* bW = (const float*)d_in[5],  *bU = (const float*)d_in[6],  *bb = (const float*)d_in[7];
  const float* sW = (const float*)d_in[8],  *sU = (const float*)d_in[9],  *sb = (const float*)d_in[10];
  const float* rW = (const float*)d_in[11], *rU = (const float*)d_in[12], *rb = (const float*)d_in[13];
  const float* Wc = (const float*)d_in[14], *bc = (const float*)d_in[15];
  const float* Wp = (const float*)d_in[16], *bp = (const float*)d_in[17];

  char* ws = (char*)d_ws;
  size_t off = 0;
  auto alloc = [&](size_t bytes){ size_t o = off; off = (off + bytes + 255) & ~(size_t)255; return o; };
  const size_t SM = alloc((size_t)B_ * S_ * H_ * 2);   // f16 stacks
  const size_t BM = alloc((size_t)B_ * S_ * H_ * 2);
  const size_t TM = alloc((size_t)B_ * S_ * 4);
  const size_t AH = alloc((size_t)B_ * H_ * 4);
  size_t PK[10];
  for (int i = 0; i < 9; ++i) PK[i] = alloc((size_t)H_ * H3_ * 2);   // 8 GRU mats + Wc
  PK[9] = alloc((size_t)H_ * V_ * 2);                                 // Wp
  (void)ws_size; (void)in_sizes; (void)n_in; (void)out_size;

  const float* g[8] = {aW, aU, bW, bU, sW, sU, rW, rU};
  for (int i = 0; i < 8; ++i)
    pack_weights<<<(H_ * H3_ + 255) / 256, 256, 0, stream>>>(g[i], (_Float16*)(ws + PK[i]), H_, H3_);
  pack_weights<<<(H3_ * H_ + 255) / 256, 256, 0, stream>>>(Wc, (_Float16*)(ws + PK[8]), H3_, H_);
  pack_weights<<<(H_ * V_ + 255) / 256, 256, 0, stream>>>(Wp, (_Float16*)(ws + PK[9]), H_, V_);

  rnng_main<<<1, 256, 0, stream>>>(
      actions, embed, ab, bb, sb, rb, bc, bp,
      (const _Float16*)(ws + PK[0]), (const _Float16*)(ws + PK[1]),
      (const _Float16*)(ws + PK[2]), (const _Float16*)(ws + PK[3]),
      (const _Float16*)(ws + PK[4]), (const _Float16*)(ws + PK[5]),
      (const _Float16*)(ws + PK[6]), (const _Float16*)(ws + PK[7]),
      (const _Float16*)(ws + PK[8]), (const _Float16*)(ws + PK[9]),
      (_Float16*)(ws + SM), (_Float16*)(ws + BM), (int*)(ws + TM),
      (float*)(ws + AH), (float*)d_out);
}
